// E77LinearMatrixCell_34059090657896
// MI455X (gfx1250) — matrix operand order repaired and checked by reading, not compiled, not run on hardware
//
#include <hip/hip_runtime.h>
#include <hip/hip_bf16.h>
#include <math.h>

#define T_DIM 2048
#define B_DIM 16
#define D_DIM 1024
#define N_DIM 64
#define M_TOTAL (T_DIM * B_DIM)   // 32768 rows
#define NCOLS   (4 * N_DIM)       // 256 projection columns
#define EPS 1e-6f

typedef __attribute__((ext_vector_type(16))) __bf16 v16bf;
typedef __attribute__((ext_vector_type(8)))  __bf16 v8bf;
typedef __attribute__((ext_vector_type(8)))  float  v8f;

__device__ inline __bf16 f32_to_bf16_rne(float f) {
    union { float f; unsigned u; } x; x.f = f;
    unsigned r = (x.u + 0x7FFFu + ((x.u >> 16) & 1u)) >> 16;
    union { unsigned short s; __bf16 b; } y; y.s = (unsigned short)r;
    return y.b;
}

// ---------------------------------------------------------------------------
// Kernel 0: streaming f32 -> bf16 conversion (8 elements / thread, b128 store).
// Pure bandwidth; keeps all conversion VALU out of the GEMM inner loop.
// ---------------------------------------------------------------------------
__global__ __launch_bounds__(256) void convert_f32_bf16(
    const float* __restrict__ src, __bf16* __restrict__ dst)
{
    const size_t base = ((size_t)blockIdx.x * 256 + threadIdx.x) * 8;
    const float4 f0 = *(const float4*)(src + base);
    const float4 f1 = *(const float4*)(src + base + 4);
    v8bf o;
    o[0] = f32_to_bf16_rne(f0.x); o[1] = f32_to_bf16_rne(f0.y);
    o[2] = f32_to_bf16_rne(f0.z); o[3] = f32_to_bf16_rne(f0.w);
    o[4] = f32_to_bf16_rne(f1.x); o[5] = f32_to_bf16_rne(f1.y);
    o[6] = f32_to_bf16_rne(f1.z); o[7] = f32_to_bf16_rne(f1.w);
    *(v8bf*)(dst + base) = o;
}

// ---------------------------------------------------------------------------
// Kernel 1: proj = x_bf16 . W_bf16^T  via v_wmma_f32_16x16x32_bf16.
// One block per 16-row M-tile; 8 waves cover the 16 N-tiles (2 each, shared A).
// Inner loop: 6x global_load_b128 + 2x WMMA. No conversion math.
// ---------------------------------------------------------------------------
__global__ __launch_bounds__(256) void proj_gemm_wmma(
    const __bf16* __restrict__ xb,   // [M_TOTAL, D_DIM]
    const __bf16* __restrict__ wb,   // [NCOLS, D_DIM]
    float* __restrict__ proj)        // [M_TOTAL, NCOLS]
{
    const int lane = threadIdx.x & 31;
    const int wave = threadIdx.x >> 5;
    const int mt   = blockIdx.x;        // 0..2047
    const int nt0  = wave * 2;          // this wave's two N-tiles

    const int half = lane >> 4;         // 0 or 1
    const int l16  = lane & 15;

    // A fragment (16x32 bf16): lane l<16 holds M=l, K in {0..7, 16..23};
    // upper half-lanes hold K in {8..15, 24..31}.
    const __bf16* arow  = xb + (size_t)(mt * 16 + l16) * D_DIM + half * 8;
    // B fragment (32x16 bf16): lane n holds col N=n, K in {0..7, 16..23} (upper: {8..15, 24..31}).
    const __bf16* brow0 = wb + (size_t)(nt0 * 16 + l16) * D_DIM + half * 8;
    const __bf16* brow1 = brow0 + (size_t)16 * D_DIM;

    v8f acc0 = {}, acc1 = {};
    for (int kk = 0; kk < D_DIM; kk += 32) {
        if (kk + 512 < D_DIM)
            __builtin_prefetch(arow + kk + 512, 0, 1);   // global_prefetch_b8

        const v8bf a_lo = *(const v8bf*)(arow + kk);
        const v8bf a_hi = *(const v8bf*)(arow + kk + 16);
        const v16bf a = __builtin_shufflevector(a_lo, a_hi,
            0,1,2,3,4,5,6,7,8,9,10,11,12,13,14,15);

        const v8bf b0l = *(const v8bf*)(brow0 + kk);
        const v8bf b0h = *(const v8bf*)(brow0 + kk + 16);
        const v16bf b0 = __builtin_shufflevector(b0l, b0h,
            0,1,2,3,4,5,6,7,8,9,10,11,12,13,14,15);

        const v8bf b1l = *(const v8bf*)(brow1 + kk);
        const v8bf b1h = *(const v8bf*)(brow1 + kk + 16);
        const v16bf b1 = __builtin_shufflevector(b1l, b1h,
            0,1,2,3,4,5,6,7,8,9,10,11,12,13,14,15);

        acc0 = __builtin_amdgcn_wmma_f32_16x16x32_bf16(
            false, a, false, b0, (short)0, acc0, false, false);
        acc1 = __builtin_amdgcn_wmma_f32_16x16x32_bf16(
            false, a, false, b1, (short)0, acc1, false, false);
    }

    // C/D layout: VGPR r -> M = r + 8*half, N = l16
    float* p0 = proj + (size_t)(mt * 16) * NCOLS + nt0 * 16 + l16;
    float* p1 = p0 + 16;
#pragma unroll
    for (int r = 0; r < 8; ++r) {
        p0[(size_t)(r + half * 8) * NCOLS] = acc0[r];
        p1[(size_t)(r + half * 8) * NCOLS] = acc1[r];
    }
}

// ---------------------------------------------------------------------------
// Kernel 2: sequential delta-rule scan. One block per batch b, 64 threads.
// Thread i owns row i of the 64x64 state S in registers.
// ---------------------------------------------------------------------------
__global__ __launch_bounds__(64) void delta_scan(
    const float* __restrict__ proj,    // [M_TOTAL, 256]  (k|v|q|g)
    const float* __restrict__ S0,      // [B, 64, 64]
    const float* __restrict__ b_gate,  // [64]
    float* __restrict__ out,           // [T, B, 64]
    float* __restrict__ Sfin)          // [B, 64, 64]
{
    const int b    = blockIdx.x;
    const int i    = threadIdx.x;     // state row 0..63
    const int lane = i & 31;
    const int wv   = i >> 5;

    __shared__ float kn_sh[N_DIM];
    __shared__ float q_sh[N_DIM];
    __shared__ float red[2];

    float S[N_DIM];
#pragma unroll
    for (int j = 0; j < N_DIM; ++j)
        S[j] = S0[((size_t)b * N_DIM + i) * N_DIM + j];

    const float bg = b_gate[i];

    for (int t = 0; t < T_DIM; ++t) {
        const float* p = proj + ((size_t)t * B_DIM + b) * NCOLS;
        const float kv = p[i];
        const float vv = p[N_DIM + i];
        const float qv = p[2 * N_DIM + i];
        const float gv = p[3 * N_DIM + i];

        // ||k||: wave32 butterfly reduce of k^2, then cross-wave via LDS
        float s = kv * kv;
#pragma unroll
        for (int off = 16; off > 0; off >>= 1)
            s += __shfl_xor(s, off, 32);
        if (lane == 0) red[wv] = s;
        __syncthreads();
        const float norm = sqrtf(red[0] + red[1]) + EPS;

        kn_sh[i] = kv / norm;
        q_sh[i]  = qv;
        __syncthreads();

        // retrieved_i = S[i,:] . k_norm
        float retr = 0.f;
#pragma unroll
        for (int j = 0; j < N_DIM; ++j) retr += S[j] * kn_sh[j];

        const float delta = vv - retr;
        const float decay = 1.f / (1.f + __expf(-(gv + bg)));

        // S[i,:] = decay_i * S[i,:] + delta_i * k_norm
#pragma unroll
        for (int j = 0; j < N_DIM; ++j) S[j] = decay * S[j] + delta * kn_sh[j];

        // Sq_i = S[i,:] . q ;  out = Sq * silu(Sq)
        float Sq = 0.f;
#pragma unroll
        for (int j = 0; j < N_DIM; ++j) Sq += S[j] * q_sh[j];
        const float sig = 1.f / (1.f + __expf(-Sq));
        out[((size_t)t * B_DIM + b) * N_DIM + i] = Sq * (Sq * sig);

        __syncthreads();   // protect kn_sh/q_sh/red before next step overwrites
    }

#pragma unroll
    for (int j = 0; j < N_DIM; ++j)
        Sfin[((size_t)b * N_DIM + i) * N_DIM + j] = S[j];
}

// ---------------------------------------------------------------------------
extern "C" void kernel_launch(void* const* d_in, const int* in_sizes, int n_in,
                              void* d_out, int out_size, void* d_ws, size_t ws_size,
                              hipStream_t stream) {
    const float* x   = (const float*)d_in[0];   // [T,B,D]
    const float* S0  = (const float*)d_in[1];   // [B,N,N]
    const float* W   = (const float*)d_in[2];   // [4N,D]
    const float* bg  = (const float*)d_in[3];   // [N]

    float* out  = (float*)d_out;                          // [T,B,N]
    float* Sfin = out + (size_t)T_DIM * B_DIM * N_DIM;    // [B,N,N]

    // Workspace layout: proj (32 MB) | x_bf16 (64 MB) | W_bf16 (0.5 MB)
    float*  proj = (float*)d_ws;
    __bf16* xb   = (__bf16*)((char*)d_ws + (size_t)M_TOTAL * NCOLS * sizeof(float));
    __bf16* wbuf = xb + (size_t)M_TOTAL * D_DIM;

    // 0) f32 -> bf16 conversion passes (streaming, bandwidth-bound)
    const size_t nx = (size_t)M_TOTAL * D_DIM;   // 33,554,432
    const size_t nw = (size_t)NCOLS * D_DIM;     //    262,144
    convert_f32_bf16<<<(unsigned)(nx / (256 * 8)), 256, 0, stream>>>(x, xb);
    convert_f32_bf16<<<(unsigned)(nw / (256 * 8)), 256, 0, stream>>>(W, wbuf);

    // 1) WMMA GEMM: one block per 16-row tile, 8 waves cover all 16 N-tiles
    proj_gemm_wmma<<<M_TOTAL / 16, 256, 0, stream>>>(xb, wbuf, proj);

    // 2) Sequential scan: one block per batch
    delta_scan<<<B_DIM, 64, 0, stream>>>(proj, S0, bg, out, Sfin);
}
